// CharRNN_55190329753724
// MI455X (gfx1250) — matrix operand order repaired and checked by reading, not compiled, not run on hardware
//
#include <hip/hip_runtime.h>
#include <math.h>

#define SEQ   256
#define BATCH 128
#define EMBD  512
#define HID   1024
#define NC    128
#define LROW  (HID + 8)   // padded LDS row (bf16 elems) -> 2064B stride, conflict-free

typedef __attribute__((ext_vector_type(16))) __bf16 v16bf;
typedef __attribute__((ext_vector_type(8)))  __bf16 v8bf;
typedef __attribute__((ext_vector_type(8)))  float  v8f;
typedef int v4i __attribute__((vector_size(16)));

union ABu { v16bf v; v8bf h[2]; };

#if defined(__HIP_DEVICE_COMPILE__) && \
    __has_builtin(__builtin_amdgcn_global_load_async_to_lds_b128) && \
    __has_builtin(__builtin_amdgcn_s_wait_asynccnt)
#define ASYNC_LDS 1
#else
#define ASYNC_LDS 0
#endif

#if defined(__HIP_DEVICE_COMPILE__) && __has_builtin(__builtin_amdgcn_sched_group_barrier)
#define SGB(mask, cnt, id) __builtin_amdgcn_sched_group_barrier((mask), (cnt), (id))
#else
#define SGB(mask, cnt, id)
#endif

// A fragment (16x32 bf16) from an LDS tile, ISA layout:
// lanes 0-15: M=lane, K = [0..7],[16..23]; lanes 16-31: M=lane-16, K=[8..15],[24..31]
__device__ __forceinline__ v16bf load_a_lds(const __bf16* hbase, int ldrow, int kc, int lane) {
  const int m  = lane & 15;
  const int c0 = (lane >> 4) << 3;            // 0 or 8
  const __bf16* p = hbase + m * ldrow + kc + c0;
  ABu u;
  u.h[0] = *(const v8bf*)(p);
  u.h[1] = *(const v8bf*)(p + 16);
  return u.v;
}

// A fragment from a global row pointer (gathered embedding row)
__device__ __forceinline__ v16bf load_a_row(const __bf16* arow, int kc, int lane) {
  const int c0 = (lane >> 4) << 3;
  const __bf16* p = arow + kc + c0;
  ABu u;
  u.h[0] = *(const v8bf*)(p);
  u.h[1] = *(const v8bf*)(p + 16);
  return u.v;
}

// B fragment (32x16 bf16): lane holds column n=lane%16, K = kc + (lane/16)*8 + [0..7],[16..23]
// B[k][n] = W[n_global][k] (W row-major [N][K]) -> two 16-byte chunks (32 bytes apart) of one W row.
__device__ __forceinline__ v16bf load_b_glb(const __bf16* wbase, int row, int ld, int kc, int lane) {
  const __bf16* p = wbase + (size_t)row * ld + kc + ((lane >> 4) << 3);
  ABu u;
  u.h[0] = *(const v8bf*)(p);
  u.h[1] = *(const v8bf*)(p + 16);
  return u.v;
}

#define WMMA_BF16(a, b, c) \
  __builtin_amdgcn_wmma_f32_16x16x32_bf16(false, (a), false, (b), (short)0, (c), false, false)

// ---------------------------------------------------------------------------
// Kernel 0: convert all weights f32 -> bf16 (one pass, segmented)
// ---------------------------------------------------------------------------
__global__ void cvt_weights_kernel(const float* __restrict__ waa,
                                   const float* __restrict__ wax,
                                   const float* __restrict__ wya,
                                   const float* __restrict__ emb,
                                   __bf16* __restrict__ owaa,
                                   __bf16* __restrict__ owax,
                                   __bf16* __restrict__ owya,
                                   __bf16* __restrict__ oemb) {
  const int i = blockIdx.x * blockDim.x + threadIdx.x;
  if (i < HID * HID)  owaa[i] = (__bf16)waa[i];
  if (i < HID * EMBD) owax[i] = (__bf16)wax[i];
  if (i < NC * HID)   owya[i] = (__bf16)wya[i];
  if (i < NC * EMBD)  oemb[i] = (__bf16)emb[i];
}

// ---------------------------------------------------------------------------
// Kernel 1: xw[s*B+b][h] = emb[seq[s][b]] @ wax^T + b_ax   (stored bf16)
// One WG = one 16-row M tile of the flattened (s,b) axis; 8 waves cover N=1024.
// Phase pipeline: 4 WMMAs per phase, next phase's 4 B-fragments in flight.
// ---------------------------------------------------------------------------
__global__ void xw_gemm_kernel(const int*    __restrict__ seq,
                               const __bf16* __restrict__ embb,
                               const __bf16* __restrict__ waxb,
                               const float*  __restrict__ b_ax,
                               __bf16*       __restrict__ xw) {
  const int tile = blockIdx.x;                 // 2048 tiles of 16 rows
  const int wave = threadIdx.x >> 5;
  const int lane = threadIdx.x & 31;
  const int m    = lane & 15;
  const int row  = tile * 16 + m;
  const int ch   = seq[row];                   // per-lane embedding gather
  const __bf16* arow = embb + (size_t)ch * EMBD;

  v8f acc[8] = {};
  v16bf aA[2];
  v16bf bb[2][4];
  aA[0] = load_a_row(arow, 0, lane);
#pragma unroll
  for (int n = 0; n < 4; ++n)
    bb[0][n] = load_b_glb(waxb, wave * 128 + n * 16 + m, EMBD, 0, lane);

  const int NPH = (EMBD / 32) * 2;             // 32 phases
#pragma unroll
  for (int p = 0; p < NPH; ++p) {
    const int c    = p >> 1;
    const int half = p & 1;
    const int cur  = p & 1;
    const int nxt  = cur ^ 1;
    if (p + 1 < NPH) {
      const int pc = (p + 1) >> 1;
      const int ph = (p + 1) & 1;
      if (ph == 0)
        aA[pc & 1] = load_a_row(arow, pc * 32, lane);
#pragma unroll
      for (int n = 0; n < 4; ++n)
        bb[nxt][n] = load_b_glb(waxb, wave * 128 + (ph * 4 + n) * 16 + m, EMBD, pc * 32, lane);
    }
#pragma unroll
    for (int n = 0; n < 4; ++n)
      acc[half * 4 + n] = WMMA_BF16(aA[c & 1], bb[cur][n], acc[half * 4 + n]);
    if (half == 1) { SGB(0x020, 2, 0); }       // next-A loads (VMEM here)
    SGB(0x020, 2, 0); SGB(0x008, 1, 0);
    SGB(0x020, 2, 0); SGB(0x008, 1, 0);
    SGB(0x020, 2, 0); SGB(0x008, 1, 0);
    SGB(0x020, 2, 0); SGB(0x008, 1, 0);
  }

#pragma unroll
  for (int n = 0; n < 8; ++n) {
    const int c    = wave * 128 + n * 16 + m;
    const float bias = b_ax[c];
#pragma unroll
    for (int r = 0; r < 8; ++r) {
      const int mrow = tile * 16 + r + ((lane >> 4) << 3);
      xw[(size_t)mrow * HID + c] = (__bf16)(acc[n][r] + bias);
    }
  }
}

// ---------------------------------------------------------------------------
// Kernel 2: persistent recurrent scan + fused output projection + h_last.
// 8 WGs, each owns a 16-row batch tile; h kept in LDS for all 256 steps.
// xw_t tiles double-buffered into LDS via ASYNC global->LDS copies (ASYNCcnt);
// waa B-fragments phase-pipelined in VGPRs (bb[2][4], no spills).
// ---------------------------------------------------------------------------
__device__ __forceinline__ void async_copy_xw(const __bf16* __restrict__ g,
                                              __bf16* l, int tid) {
#if ASYNC_LDS
#pragma unroll
  for (int j = 0; j < 8; ++j) {
    const int e = (j * 256 + tid) * 8;        // 8 bf16 = 16B per async op
    __builtin_amdgcn_global_load_async_to_lds_b128(
        (__attribute__((address_space(1))) v4i*)(g + e),
        (__attribute__((address_space(3))) v4i*)(l + e), 0, 0);
  }
#else
  (void)g; (void)l; (void)tid;
#endif
}

__global__ void rnn_scan_kernel(const __bf16* __restrict__ waab,
                                const __bf16* __restrict__ wyab,
                                const __bf16* __restrict__ xw,
                                const float*  __restrict__ b_ya,
                                float*        __restrict__ d_out) {
  __shared__ __bf16 h_lds[16 * LROW];          // ~33 KB, bank-conflict padded
#if ASYNC_LDS
  __shared__ __bf16 xwb_lds[2][16 * HID];      // 64 KB double buffer
#endif

  const int tid  = threadIdx.x;
  const int b0   = blockIdx.x * 16;
  const int wave = tid >> 5;
  const int lane = tid & 31;
  const int m    = lane & 15;
  const int rsel = (lane >> 4) << 3;           // row offset 0/8 for C fragments

  for (int i = tid; i < 16 * LROW; i += blockDim.x) h_lds[i] = (__bf16)0.0f;

#if ASYNC_LDS
  async_copy_xw(xw + (size_t)b0 * HID, &xwb_lds[0][0], tid);   // preload t=0
#endif
  __syncthreads();

  for (int t = 0; t < SEQ; ++t) {
#if ASYNC_LDS
    if (t + 1 < SEQ)                            // stream next xw tile under the GEMM
      async_copy_xw(xw + ((size_t)(t + 1) * BATCH + b0) * HID,
                    &xwb_lds[(t + 1) & 1][0], tid);
#else
    const __bf16* xwt_g = xw + ((size_t)t * BATCH + b0) * HID;
    __builtin_prefetch(xwt_g, 0, 1);
#endif

    // ---- recurrent GEMM: acc[n] = h @ waa^T  (wave covers cols wave*128..+127)
    // Phase pipeline: 4 WMMAs per phase with the other 4-frag buffer in flight.
    v8f acc[8] = {};
    v16bf aA[2];
    v16bf bb[2][4];
    aA[0] = load_a_lds(h_lds, LROW, 0, lane);
#pragma unroll
    for (int n = 0; n < 4; ++n)
      bb[0][n] = load_b_glb(waab, wave * 128 + n * 16 + m, HID, 0, lane);

    const int NPH = (HID / 32) * 2;             // 64 phases
#pragma unroll
    for (int p = 0; p < NPH; ++p) {
      const int c    = p >> 1;
      const int half = p & 1;
      const int cur  = p & 1;
      const int nxt  = cur ^ 1;
      if (p + 1 < NPH) {
        const int pc = (p + 1) >> 1;
        const int ph = (p + 1) & 1;
        if (ph == 0)
          aA[pc & 1] = load_a_lds(h_lds, LROW, pc * 32, lane);
#pragma unroll
        for (int n = 0; n < 4; ++n)
          bb[nxt][n] = load_b_glb(waab, wave * 128 + (ph * 4 + n) * 16 + m, HID, pc * 32, lane);
      }
#pragma unroll
      for (int n = 0; n < 4; ++n)
        acc[half * 4 + n] = WMMA_BF16(aA[c & 1], bb[cur][n], acc[half * 4 + n]);
      if (half == 1) { SGB(0x100, 2, 0); }     // next-A (DS reads)
      SGB(0x020, 2, 0); SGB(0x008, 1, 0);
      SGB(0x020, 2, 0); SGB(0x008, 1, 0);
      SGB(0x020, 2, 0); SGB(0x008, 1, 0);
      SGB(0x020, 2, 0); SGB(0x008, 1, 0);
    }

#if ASYNC_LDS
    // current buffer's 8 per-thread async ops are the oldest outstanding:
    // <=8 remaining  =>  current buffer complete (in-order completion).
    if (t + 1 < SEQ) __builtin_amdgcn_s_wait_asynccnt(8);
    else             __builtin_amdgcn_s_wait_asynccnt(0);
#endif
    __syncthreads();                            // reads of h_t done + xw_t visible

    // ---- h_{t+1} = tanh(acc + xw_t), written back to LDS
#if ASYNC_LDS
    const __bf16* xwt = &xwb_lds[t & 1][0];
#else
    const __bf16* xwt = xwt_g;
#endif
#pragma unroll
    for (int n = 0; n < 8; ++n) {
      const int col = wave * 128 + n * 16 + m;
#pragma unroll
      for (int r = 0; r < 8; ++r) {
        const int mr = r + rsel;
        const float v = acc[n][r] + (float)xwt[mr * HID + col];
        h_lds[mr * LROW + col] = (__bf16)tanhf(v);
      }
    }
    __syncthreads();                            // h_{t+1} visible

    // ---- fused output projection: out_t = h_{t+1} @ wya^T + b_ya
    // NC = 128 -> one 16-col tile per wave; software-pipelined K loop.
    v8f oacc = {};
    v16bf oa = load_a_lds(h_lds, LROW, 0, lane);
    v16bf ob = load_b_glb(wyab, wave * 16 + m, HID, 0, lane);
#pragma unroll
    for (int kc = 0; kc < HID; kc += 32) {
      v16bf na = oa, nb = ob;
      if (kc + 32 < HID) {
        na = load_a_lds(h_lds, LROW, kc + 32, lane);
        nb = load_b_glb(wyab, wave * 16 + m, HID, kc + 32, lane);
      }
      oacc = WMMA_BF16(oa, ob, oacc);
      oa = na; ob = nb;
      SGB(0x100, 2, 0);   // next A (DS)
      SGB(0x020, 2, 0);   // next B (VMEM)
      SGB(0x008, 1, 0);   // current WMMA
    }
    const int c = wave * 16 + m;
    const float bias = b_ya[c];
    float* op = d_out + ((size_t)t * BATCH + b0) * NC;
#pragma unroll
    for (int r = 0; r < 8; ++r) {
      op[(r + rsel) * NC + c] = oacc[r] + bias;
    }
    // next iteration's post-read barrier protects these LDS reads vs. future writes
  }

  __syncthreads();
  // ---- h_last [BATCH][HID] appended after out in d_out
  float* hl = d_out + (size_t)SEQ * BATCH * NC + (size_t)b0 * HID;
  for (int i = tid; i < 16 * HID; i += blockDim.x) {
    const int mr = i / HID, cc = i % HID;
    hl[(size_t)mr * HID + cc] = (float)h_lds[mr * LROW + cc];
  }
}

// ---------------------------------------------------------------------------
extern "C" void kernel_launch(void* const* d_in, const int* in_sizes, int n_in,
                              void* d_out, int out_size, void* d_ws, size_t ws_size,
                              hipStream_t stream) {
  const int*   input_seq = (const int*)  d_in[0];
  const float* emb       = (const float*)d_in[1];
  const float* waa       = (const float*)d_in[2];
  const float* wax       = (const float*)d_in[3];
  const float* b_ax      = (const float*)d_in[4];
  const float* wya       = (const float*)d_in[5];
  const float* b_ya      = (const float*)d_in[6];
  float* out = (float*)d_out;

  // workspace layout (bytes)
  char* ws = (char*)d_ws;
  __bf16* waab = (__bf16*)(ws);                                       // 2 MB
  __bf16* waxb = (__bf16*)(ws + 2097152);                             // 1 MB
  __bf16* wyab = (__bf16*)(ws + 2097152 + 1048576);                   // 256 KB
  __bf16* embb = (__bf16*)(ws + 2097152 + 1048576 + 262144);          // 128 KB
  __bf16* xwb  = (__bf16*)(ws + 2097152 + 1048576 + 262144 + 131072); // 64 MB

  // 0) weight conversion f32 -> bf16
  cvt_weights_kernel<<<(HID * HID + 255) / 256, 256, 0, stream>>>(
      waa, wax, wya, emb, waab, waxb, wyab, embb);

  // 1) xw = emb[seq] @ wax^T + b_ax  (bf16), 2048 M-tiles
  xw_gemm_kernel<<<(SEQ * BATCH) / 16, 256, 0, stream>>>(
      input_seq, embb, waxb, b_ax, xwb);

  // 2) persistent scan: 8 WGs x 256 threads, fused output projection
  rnn_scan_kernel<<<BATCH / 16, 256, 0, stream>>>(
      waab, wyab, xwb, b_ya, out);

  (void)in_sizes; (void)n_in; (void)out_size; (void)ws_size;
}
